// EnerG_54889682043542
// MI455X (gfx1250) — matrix operand order repaired and checked by reading, not compiled, not run on hardware
//
#include <hip/hip_runtime.h>

typedef __attribute__((ext_vector_type(16))) __bf16 v16bf;
typedef __attribute__((ext_vector_type(8)))  float  v8f;

#define LEAKY(v) fmaxf((v), 0.1f*(v))

union AFrag { unsigned int u[8]; v16bf v; };

// ---- CDNA5 async global->LDS helpers (ASYNCcnt-tracked, no VGPR data path)
// GVS form: global_load_async_to_lds_b64 vLdsAddr, vOffset, sBase
__device__ __forceinline__ void async_stage_2kb(const unsigned int* __restrict__ gbase,
                                                void* ldsdst) {
  unsigned lds  = (unsigned)(unsigned long long)ldsdst + threadIdx.x * 8u;
  unsigned voff = threadIdx.x * 8u;
  asm volatile("global_load_async_to_lds_b64 %0, %1, %2"
               :: "v"(lds), "v"(voff), "s"(gbase) : "memory");
}
__device__ __forceinline__ void wait_async0() {
#if __has_builtin(__builtin_amdgcn_s_wait_asynccnt)
  __builtin_amdgcn_s_wait_asynccnt(0);
#else
  asm volatile("s_wait_asynccnt 0x0" ::: "memory");
#endif
}

// ---------------------------------------------------------------- zero
__global__ void k_zero(float* __restrict__ p, long long n) {
  long long i = (long long)blockIdx.x * blockDim.x + threadIdx.x;
  long long st = (long long)gridDim.x * blockDim.x;
  for (; i < n; i += st) p[i] = 0.f;
}

// ------------------------------------------- pack w2 into WMMA-B lane image
// Packed dword index: ct*512 + c*256 + lane*8 + v
// half j of dword: K = 32*c + 16*(lane>>4) + 2*v + j ; N = lane&15 ; col = ct*16+N
__global__ void k_pack_w2(const float* __restrict__ w2, unsigned int* __restrict__ pk,
                          int inout, int ndw) {
  int i = blockIdx.x * blockDim.x + threadIdx.x;
  int st = gridDim.x * blockDim.x;
  for (; i < ndw; i += st) {
    int v = i & 7, lane = (i >> 3) & 31, c = (i >> 8) & 1, ct = i >> 9;
    int hh = lane >> 4, nn = lane & 15;
    int K0 = 32 * c + 16 * hh + 2 * v;
    int col = ct * 16 + nn;
    union { __bf16 h[2]; unsigned int u; } t;
    t.h[0] = (__bf16)w2[(size_t)K0 * inout + col];
    t.h[1] = (__bf16)w2[(size_t)(K0 + 1) * inout + col];
    pk[i] = t.u;
  }
}

// --------------------------------------------- edge features for all 3 nets
__global__ void k_edge_feat(const float* __restrict__ x,
    const int* __restrict__ src, const int* __restrict__ dst,
    const float* __restrict__ w1a, const float* __restrict__ b1a,
    const float* __restrict__ w1b, const float* __restrict__ b1b,
    const float* __restrict__ w1c, const float* __restrict__ b1c,
    __bf16* __restrict__ e1, __bf16* __restrict__ e2, __bf16* __restrict__ e3,
    int E) {
  int e = blockIdx.x * blockDim.x + threadIdx.x;
  if (e >= E) return;
  int s = src[e], d = dst[e];
  float a0 = x[d*4+1] - x[s*4+1];
  float a1 = x[d*4+2] - x[s*4+2];
  float a2 = x[d*4+3] - x[s*4+3];
  #pragma unroll 4
  for (int o = 0; o < 64; o++) {
    float v;
    v = b1a[o] + a0*w1a[o] + a1*w1a[64+o] + a2*w1a[128+o];
    e1[(size_t)e*64 + o] = (__bf16)LEAKY(v);
    v = b1b[o] + a0*w1b[o] + a1*w1b[64+o] + a2*w1b[128+o];
    e2[(size_t)e*64 + o] = (__bf16)LEAKY(v);
    v = b1c[o] + a0*w1c[o] + a1*w1c[64+o] + a2*w1c[128+o];
    e3[(size_t)e*64 + o] = (__bf16)LEAKY(v);
  }
}

// --------------------------------------------- layer 1 (4->8, scalar path)
__global__ void k_layer1(const __bf16* __restrict__ e1,
    const float* __restrict__ w2, const float* __restrict__ b2,
    const float* __restrict__ x,
    const int* __restrict__ src, const int* __restrict__ dst,
    float* __restrict__ agg1, int E) {
  int e = blockIdx.x * blockDim.x + threadIdx.x;
  if (e >= E) return;
  float ef[64];
  #pragma unroll
  for (int k = 0; k < 64; k++) ef[k] = (float)e1[(size_t)e*64 + k];
  int s = src[e];
  float xs[4];
  #pragma unroll
  for (int i = 0; i < 4; i++) xs[i] = x[s*4 + i];
  float msg[8] = {0,0,0,0,0,0,0,0};
  for (int t = 0; t < 32; t++) {
    float wv = b2[t];
    #pragma unroll
    for (int k = 0; k < 64; k++) wv = fmaf(ef[k], w2[k*32 + t], wv);
    wv = LEAKY(wv);
    msg[t & 7] = fmaf(xs[t >> 3], wv, msg[t & 7]);
  }
  int d = dst[e];
  #pragma unroll
  for (int o = 0; o < 8; o++) atomicAdd(&agg1[(size_t)d*8 + o], msg[o]);
}

// --------------------------------------------- node root update (generic)
__global__ void k_node_update(const float* __restrict__ agg,
    const float* __restrict__ hin, const float* __restrict__ root,
    const float* __restrict__ cb, float* __restrict__ hout,
    int Nn, int IC, int OC) {
  int idx = blockIdx.x * blockDim.x + threadIdx.x;
  if (idx >= Nn * OC) return;
  int n = idx / OC, o = idx - n * OC;
  float v = agg[idx] + cb[o];
  for (int i = 0; i < IC; i++) v = fmaf(hin[(size_t)n*IC + i], root[i*OC + o], v);
  hout[idx] = LEAKY(v);
}

// ---------------------------------------------------------------- fused NNConv
// Per 16-edge tile (one wave): W = leaky(ef@w2+b) computed in 16x16 WMMA tiles,
// immediately contracted against h[src] and scatter-added to agg[dst].
// B tiles stream through double-buffered LDS, filled by ASYNCcnt-tracked
// global_load_async_to_lds, shared by all 8 waves of the block.
template<int IN, int OUT>
__launch_bounds__(256)
__global__ void k_nnconv_wmma(const __bf16* __restrict__ ef,
    const unsigned int* __restrict__ packedB,
    const float* __restrict__ b2,
    const float* __restrict__ h,
    const int* __restrict__ src, const int* __restrict__ dst,
    float* __restrict__ agg, int E) {
  constexpr int INOUT = IN * OUT;
  constexpr int NCT   = INOUT / 16;   // 16-wide column tiles
  constexpr int OB    = OUT / 16;     // col tiles per input channel
  __shared__ __align__(32) __bf16 sB[2][1024];  // 2 x 2KB (64K x 16N bf16)
  __shared__ float sH[8][16 * IN];

  const int wave = threadIdx.x >> 5;
  const int lane = threadIdx.x & 31;
  const int hh   = lane >> 4;
  const int m    = lane & 15;
  const int tile = blockIdx.x * 8 + wave;
  const int e0   = tile * 16;

  // stage h[src] rows for this wave's 16 edges
  {
    int e = e0 + m; if (e >= E) e = E - 1;
    int s = src[e];
    #pragma unroll
    for (int i = 0; i < IN/2; ++i)
      sH[wave][m*IN + hh*(IN/2) + i] = h[(size_t)s*IN + hh*(IN/2) + i];
  }

  // A fragments: 16 edges x 64 K (bf16), register-resident for whole tile
  // A layout (contiguous, matches packed B): K = 16h+2v+{0,1}; +32c
  AFrag A[2];
  {
    int e = e0 + m; if (e >= E) e = E - 1;
    const __bf16* arow = ef + (size_t)e * 64;
    #pragma unroll
    for (int c = 0; c < 2; c++)
      #pragma unroll
      for (int v = 0; v < 8; v++) {
        int K0 = 32*c + 16*hh + 2*v;
        A[c].u[v] = *(const unsigned int*)(arow + K0);
      }
  }

  // async-stage first B tile (2KB: one b64 per thread)
  async_stage_2kb(packedB, &sB[0][0]);
  wait_async0();
  __syncthreads();

  float acc[OB][8];
  #pragma unroll
  for (int ob = 0; ob < OB; ob++)
    #pragma unroll
    for (int r = 0; r < 8; r++) acc[ob][r] = 0.f;

  for (int i = 0; i < IN; i++) {
    float hv[8];
    #pragma unroll
    for (int r = 0; r < 8; r++) hv[r] = sH[wave][(r + 8*hh)*IN + i];
    #pragma unroll
    for (int ob = 0; ob < OB; ob++) {
      const int ct  = i * OB + ob;
      const int buf = ct & 1;
      if (ct + 1 < NCT)   // async prefetch of next tile overlaps this tile's WMMAs
        async_stage_2kb(packedB + (size_t)(ct + 1) * 512, &sB[buf ^ 1][0]);

      v16bf B0 = *(const v16bf*)&sB[buf][lane * 16];
      v16bf B1 = *(const v16bf*)&sB[buf][(32 + lane) * 16];
      v8f c = {};
      c = __builtin_amdgcn_wmma_f32_16x16x32_bf16(false, A[0].v, false, B0,
                                                  (short)0, c, false, false);
      c = __builtin_amdgcn_wmma_f32_16x16x32_bf16(false, A[1].v, false, B1,
                                                  (short)0, c, false, false);
      float bias = b2[ct * 16 + m];
      #pragma unroll
      for (int r = 0; r < 8; r++) {
        float w = c[r] + bias;
        w = LEAKY(w);
        acc[ob][r] = fmaf(hv[r], w, acc[ob][r]);
      }
      wait_async0();      // prefetched tile resident in LDS
      __syncthreads();    // all waves past reads of current buffer
    }
  }

  // scatter messages: element acc[ob][r] -> edge M=r+8*hh, out channel ob*16+m
  #pragma unroll
  for (int r = 0; r < 8; r++) {
    int e = e0 + r + 8*hh;
    if (e < E) {
      int d = dst[e];
      #pragma unroll
      for (int ob = 0; ob < OB; ob++)
        atomicAdd(&agg[(size_t)d*OUT + ob*16 + m], acc[ob][r]);
    }
  }
}

// --------------------- layer-3 node update fused with sorted sum-pool
__global__ void k_node3_pool(const float* __restrict__ agg3,
    const float* __restrict__ h2, const float* __restrict__ root3,
    const float* __restrict__ cb3, const int* __restrict__ batch,
    float* __restrict__ gpool, int N) {
  int o  = threadIdx.x;          // 0..127 out channel
  int n0 = blockIdx.x * 64;
  float run = 0.f; int curg = -1;
  for (int t = 0; t < 64; t++) {
    int n = n0 + t;
    if (n >= N) break;
    float v = agg3[(size_t)n*128 + o] + cb3[o];
    for (int i = 0; i < 64; i++) v = fmaf(h2[(size_t)n*64 + i], root3[i*128 + o], v);
    v = LEAKY(v);
    int g = batch[n];
    if (g != curg) {
      if (curg >= 0) atomicAdd(&gpool[(size_t)curg*128 + o], run);
      run = 0.f; curg = g;
    }
    run += v;
  }
  if (curg >= 0) atomicAdd(&gpool[(size_t)curg*128 + o], run);
}

// ---------------------------------------------------------------- readout MLP
__global__ void k_readout(const float* __restrict__ g,
    const float* __restrict__ w1, const float* __restrict__ b1,
    const float* __restrict__ w2, const float* __restrict__ b2,
    const float* __restrict__ w3, const float* __restrict__ b3,
    float* __restrict__ out) {
  __shared__ float sa1[32][128];
  __shared__ float sa2[32][64];
  int gi = threadIdx.x;
  if (gi >= 32) return;
  for (int o = 0; o < 128; o++) {
    float v = b1[o];
    for (int i = 0; i < 128; i++) v = fmaf(g[gi*128 + i], w1[i*128 + o], v);
    sa1[gi][o] = LEAKY(v);
  }
  for (int o = 0; o < 64; o++) {
    float v = b2[o];
    for (int i = 0; i < 128; i++) v = fmaf(sa1[gi][i], w2[i*64 + o], v);
    sa2[gi][o] = LEAKY(v);
  }
  float v = b3[0];
  for (int i = 0; i < 64; i++) v = fmaf(sa2[gi][i], w3[i], v);
  out[gi] = v;
}

// ================================================================= launcher
extern "C" void kernel_launch(void* const* d_in, const int* in_sizes, int n_in,
                              void* d_out, int out_size, void* d_ws, size_t ws_size,
                              hipStream_t stream) {
  const float* x      = (const float*)d_in[0];
  const int*   ei     = (const int*)  d_in[1];
  const int*   batch  = (const int*)  d_in[2];
  const float* en1_w1 = (const float*)d_in[3];
  const float* en1_b1 = (const float*)d_in[4];
  const float* en1_w2 = (const float*)d_in[5];
  const float* en1_b2 = (const float*)d_in[6];
  const float* en2_w1 = (const float*)d_in[7];
  const float* en2_b1 = (const float*)d_in[8];
  const float* en2_w2 = (const float*)d_in[9];
  const float* en2_b2 = (const float*)d_in[10];
  const float* en3_w1 = (const float*)d_in[11];
  const float* en3_b1 = (const float*)d_in[12];
  const float* en3_w2 = (const float*)d_in[13];
  const float* en3_b2 = (const float*)d_in[14];
  const float* root1  = (const float*)d_in[15];
  const float* cb1    = (const float*)d_in[16];
  const float* root2  = (const float*)d_in[17];
  const float* cb2    = (const float*)d_in[18];
  const float* root3  = (const float*)d_in[19];
  const float* cb3    = (const float*)d_in[20];
  const float* fc1_w  = (const float*)d_in[21];
  const float* fc1_b  = (const float*)d_in[22];
  const float* fc2_w  = (const float*)d_in[23];
  const float* fc2_b  = (const float*)d_in[24];
  const float* fc3_w  = (const float*)d_in[25];
  const float* fc3_b  = (const float*)d_in[26];

  const int N = in_sizes[0] / 4;
  const int E = in_sizes[1] / 2;
  const int* src = ei;
  const int* dst = ei + E;

  char* p = (char*)d_ws;
  auto carve = [&](size_t bytes) -> char* {
    char* r = p; p += (bytes + 255) & ~(size_t)255; return r;
  };
  __bf16* e1 = (__bf16*)carve((size_t)E * 64 * 2);
  __bf16* e2 = (__bf16*)carve((size_t)E * 64 * 2);
  __bf16* e3 = (__bf16*)carve((size_t)E * 64 * 2);
  unsigned int* pk2 = (unsigned int*)carve((size_t)64 * 512  * 2);
  unsigned int* pk3 = (unsigned int*)carve((size_t)64 * 8192 * 2);
  float* h1 = (float*)carve((size_t)N * 8  * 4);
  float* h2 = (float*)carve((size_t)N * 64 * 4);
  char* zbeg = p;                              // zeroed region (accumulators)
  float* agg1  = (float*)carve((size_t)N * 8   * 4);
  float* agg2  = (float*)carve((size_t)N * 64  * 4);
  float* agg3  = (float*)carve((size_t)N * 128 * 4);
  float* gpool = (float*)carve((size_t)32 * 128 * 4);
  long long nz = (long long)(p - zbeg) / 4;

  k_zero<<<512, 256, 0, stream>>>((float*)zbeg, nz);

  int ndw2 = 64 * 512 / 2;
  k_pack_w2<<<(ndw2 + 255) / 256, 256, 0, stream>>>(en2_w2, pk2, 512, ndw2);
  int ndw3 = 64 * 8192 / 2;
  k_pack_w2<<<(ndw3 + 255) / 256, 256, 0, stream>>>(en3_w2, pk3, 8192, ndw3);

  k_edge_feat<<<(E + 255) / 256, 256, 0, stream>>>(x, src, dst,
      en1_w1, en1_b1, en2_w1, en2_b1, en3_w1, en3_b1, e1, e2, e3, E);

  k_layer1<<<(E + 255) / 256, 256, 0, stream>>>(e1, en1_w2, en1_b2,
      x, src, dst, agg1, E);
  k_node_update<<<(N * 8 + 255) / 256, 256, 0, stream>>>(agg1, x, root1, cb1,
      h1, N, 4, 8);

  int tiles  = (E + 15) / 16;
  int blocks = (tiles + 7) / 8;
  k_nnconv_wmma<8, 64><<<blocks, 256, 0, stream>>>(e2, pk2, en2_b2, h1,
      src, dst, agg2, E);
  k_node_update<<<(N * 64 + 255) / 256, 256, 0, stream>>>(agg2, h1, root2, cb2,
      h2, N, 8, 64);

  k_nnconv_wmma<64, 128><<<blocks, 256, 0, stream>>>(e3, pk3, en3_b2, h2,
      src, dst, agg3, E);

  k_node3_pool<<<(N + 63) / 64, 128, 0, stream>>>(agg3, h2, root3, cb3, batch,
      gpool, N);
  k_readout<<<1, 32, 0, stream>>>(gpool, fc1_w, fc1_b, fc2_w, fc2_b,
      fc3_w, fc3_b, (float*)d_out);
}
